// Net_32882269618241
// MI455X (gfx1250) — matrix operand order repaired and checked by reading, not compiled, not run on hardware
//
#include <hip/hip_runtime.h>
#include <cstdint>
#include <cstddef>

// ---------------------------------------------------------------- constants
#define HDIM   577            // LSTM hidden size
#define HP     608            // HDIM padded to 19*32 (WMMA K granularity)
#define TSTEPS 512
#define BATCH  64
#define NACT   64
#define JT     38             // HP/16 j-tiles per layer (one per workgroup)
#define KSX    19             // K-steps (of 32) covering one HP-wide input half
#define KS2    38             // K-steps covering [x ; h] = 1216
#define NWG    38
#define NTHREADS 256

// d_out layout (floats): logits | baseline | action | hT | cT
#define LOGITS_N (TSTEPS*BATCH*NACT)          // 2097152
#define BASE_OFF (LOGITS_N)
#define ACT_OFF  (BASE_OFF + TSTEPS*BATCH)
#define HT_OFF   (ACT_OFF  + TSTEPS*BATCH)
#define CT_OFF   (HT_OFF   + 2*BATCH*HDIM)

typedef __attribute__((ext_vector_type(16))) __bf16 v16bf;
typedef __attribute__((ext_vector_type(8)))  float  v8f;

union Frag {                       // one WMMA operand per lane: 16 bf16 = 32B
  v16bf v;
  uint4 u[2];
};

__device__ __forceinline__ float sigm(float x) { return 1.0f / (1.0f + __expf(-x)); }

__device__ __forceinline__ v8f wmma_bf16(const Frag& a, const Frag& b, v8f c) {
  // D(f32 16x16) = A(bf16 16x32) * B(bf16 32x16) + C
  return __builtin_amdgcn_wmma_f32_16x16x32_bf16(false, a.v, false, b.v,
                                                 (short)0, c, false, false);
}

// sense-reversing grid barrier across the NWG persistent workgroups
__device__ __forceinline__ void grid_sync(unsigned* bar) {
  __syncthreads();
  if (threadIdx.x == 0) {
    __threadfence();
    volatile unsigned* vb = (volatile unsigned*)bar;
    unsigned gen = vb[1];
    if (atomicAdd(&bar[0], 1u) == NWG - 1u) {
      vb[0] = 0u;
      __threadfence();
      atomicAdd(&bar[1], 1u);
    } else {
      while (vb[1] == gen) { __builtin_amdgcn_s_sleep(2); }
    }
  }
  __syncthreads();
}

// ---------------------------------------------------------------- prep kernels
__global__ void k_init(unsigned* bar) {
  if (threadIdx.x < 2) bar[threadIdx.x] = 0u;
}

// core_in = [frame(512) | clip(reward)(1) | one_hot(64) | pad(31)]  -> bf16
__global__ void k_build_x(const float* __restrict__ frame,
                          const int* __restrict__ act,
                          const float* __restrict__ rew,
                          __bf16* __restrict__ xbuf) {
  size_t e = (size_t)blockIdx.x * blockDim.x + threadIdx.x;
  if (e >= (size_t)TSTEPS * BATCH * HP) return;
  int col = (int)(e % HP);
  size_t row = e / HP;                         // t*BATCH + b
  float v = 0.f;
  if (col < 512)            v = frame[row * 512 + col];
  else if (col == 512)      { float r = rew[row]; v = fminf(1.f, fmaxf(-1.f, r)); }
  else if (col < HDIM)      v = (act[row] == (col - 513)) ? 1.f : 0.f;
  xbuf[e] = (__bf16)v;
}

// pack [W_ih^T ; W_hh^T] per (layer, j-tile, gate, k-step) into per-lane
// B-fragment order: frag = 32 lanes x 16 halves.  lane&15 -> local output
// column j (0..15), (lane>>4, half) -> k in table order.
__global__ void k_pack_w(const float* __restrict__ wih0, const float* __restrict__ whh0,
                         const float* __restrict__ wih1, const float* __restrict__ whh1,
                         __bf16* __restrict__ wpack) {
  size_t e = (size_t)blockIdx.x * blockDim.x + threadIdx.x;
  if (e >= (size_t)2 * JT * 4 * KS2 * 512) return;
  int hh = (int)(e & 15);
  int lane = (int)((e >> 4) & 31);
  size_t fragIdx = e >> 9;
  int ks = (int)(fragIdx % KS2);
  size_t q = fragIdx / KS2;
  int g = (int)(q & 3); q >>= 2;
  int jt = (int)(q % JT);
  int l  = (int)(q / JT);
  int jcol = jt * 16 + (lane & 15);
  float v = 0.f;
  if (jcol < HDIM) {
    size_t rowW = (size_t)(g * HDIM + jcol);
    if (ks < KSX) {                            // x half -> W_ih
      int k = ks * 32 + (hh & 7) + 8 * (lane >> 4) + 16 * (hh >> 3);
      if (k < HDIM) { const float* w = l ? wih1 : wih0; v = w[rowW * HDIM + k]; }
    } else {                                   // h half -> W_hh
      int k = (ks - KSX) * 32 + (hh & 7) + 8 * (lane >> 4) + 16 * (hh >> 3);
      if (k < HDIM) { const float* w = l ? whh1 : whh0; v = w[rowW * HDIM + k]; }
    }
  }
  wpack[e] = (__bf16)v;
}

__global__ void k_pack_p(const float* __restrict__ pw, __bf16* __restrict__ ppack) {
  size_t e = (size_t)blockIdx.x * blockDim.x + threadIdx.x;
  if (e >= (size_t)4 * KSX * 512) return;
  int hh = (int)(e & 15);
  int lane = (int)((e >> 4) & 31);
  size_t f = e >> 9;
  int ks = (int)(f % KSX);
  int nt = (int)(f / KSX);
  int n = nt * 16 + (lane & 15);
  int k = ks * 32 + (hh & 7) + 8 * (lane >> 4) + 16 * (hh >> 3);
  float v = (k < HDIM) ? pw[(size_t)n * HDIM + k] : 0.f;
  ppack[e] = (__bf16)v;
}

// bias[l][gate][HP] = b_ih + b_hh (fp32, padded with zeros)
__global__ void k_pack_b(const float* __restrict__ bih0, const float* __restrict__ bhh0,
                         const float* __restrict__ bih1, const float* __restrict__ bhh1,
                         float* __restrict__ bias) {
  int e = blockIdx.x * blockDim.x + threadIdx.x;
  if (e >= 2 * 4 * HP) return;
  int j = e % HP;
  int g = (e / HP) & 3;
  int l = e / (HP * 4);
  float v = 0.f;
  if (j < HDIM) {
    const float* bi = l ? bih1 : bih0;
    const float* bh = l ? bhh1 : bhh0;
    v = bi[g * HDIM + j] + bh[g * HDIM + j];
  }
  bias[e] = v;
}

__global__ void k_init_h(const float* __restrict__ h0in,
                         __bf16* __restrict__ h0buf, __bf16* __restrict__ h1buf) {
  int e = blockIdx.x * blockDim.x + threadIdx.x;
  if (e >= 2 * BATCH * HP) return;
  int j = e % HP;
  int b = (e / HP) % BATCH;
  int l = e / (HP * BATCH);
  float v = (j < HDIM) ? h0in[((size_t)l * BATCH + b) * HDIM + j] : 0.f;
  __bf16 bv = (__bf16)v;
  if (l == 0) h0buf[(size_t)b * HP + j] = bv;   // parity 0
  else        h1buf[(size_t)b * HP + j] = bv;
}

// ---------------------------------------------------------------- LSTM core
// One gate-GEMM layer slice for this workgroup's j-tile:
//   gates(64 x 64) = [x(64xHP) ; nd*h(64xHP)] @ Wslice(1216 x 64) ,
// waves: mt = wave>>1 (batch tile), pair = wave&1 -> gates {i,f} or {g,o}.
__device__ __forceinline__ void lstm_phase(
    const __bf16* __restrict__ xb,        // x rows (stride HP)
    const __bf16* __restrict__ hb,        // h_prev rows (stride HP)
    const unsigned char* __restrict__ dn, // done + t*BATCH
    const __bf16* wl,                     // LDS weights for this layer [4][KS2][512]
    float* ifl,                           // LDS i/f staging [4][2][16][16] f32
    const float* bias4,                   // bias for this layer [4][HP]
    float* cloc,                          // per-thread c state (pair==1 only)
    __bf16* hdst, __bf16* ydst,
    float* hfin, float* cfin,             // final-step f32 outputs (or null)
    int jt, int wave, int lane) {
  const int mt = wave >> 1, pair = wave & 1;
  const int hl = lane >> 4, ml = lane & 15, jl = ml;
  const int g0 = pair * 2;
  const int hoff = 8 * hl;
  v8f a0 = {}; v8f a1 = {};
  Frag fa, f0, f1;

  // --- x half (never masked)
  for (int ks = 0; ks < KSX; ++ks) {
    const __bf16* row = xb + (size_t)(mt * 16 + ml) * HP + ks * 32;
    __builtin_prefetch(row + 32, 0, 1);
    fa.u[0] = *(const uint4*)(row + hoff);
    fa.u[1] = *(const uint4*)(row + 16 + hoff);
    const __bf16* w0 = wl + ((size_t)(g0 * KS2 + ks)) * 512 + lane * 16;
    const __bf16* w1 = wl + ((size_t)((g0 + 1) * KS2 + ks)) * 512 + lane * 16;
    f0.u[0] = *(const uint4*)w0; f0.u[1] = *(const uint4*)(w0 + 8);
    f1.u[0] = *(const uint4*)w1; f1.u[1] = *(const uint4*)(w1 + 8);
    a0 = wmma_bf16(fa, f0, a0);
    a1 = wmma_bf16(fa, f1, a1);
  }
  // --- h half (zeroed where done[t,b])
  const unsigned char d = dn[mt * 16 + ml];
  const uint4 z4 = make_uint4(0u, 0u, 0u, 0u);
  for (int ks = 0; ks < KSX; ++ks) {
    const __bf16* row = hb + (size_t)(mt * 16 + ml) * HP + ks * 32;
    fa.u[0] = *(const uint4*)(row + hoff);
    fa.u[1] = *(const uint4*)(row + 16 + hoff);
    if (d) { fa.u[0] = z4; fa.u[1] = z4; }
    const int kk = KSX + ks;
    const __bf16* w0 = wl + ((size_t)(g0 * KS2 + kk)) * 512 + lane * 16;
    const __bf16* w1 = wl + ((size_t)((g0 + 1) * KS2 + kk)) * 512 + lane * 16;
    f0.u[0] = *(const uint4*)w0; f0.u[1] = *(const uint4*)(w0 + 8);
    f1.u[0] = *(const uint4*)w1; f1.u[1] = *(const uint4*)(w1 + 8);
    a0 = wmma_bf16(fa, f0, a0);
    a1 = wmma_bf16(fa, f1, a1);
  }

  // stage i,f to LDS so the g/o waves can run the full cell locally
  if (pair == 0) {
    for (int r = 0; r < 8; ++r) {
      ifl[((mt * 2 + 0) * 16 + r + hoff) * 16 + jl] = a0[r];
      ifl[((mt * 2 + 1) * 16 + r + hoff) * 16 + jl] = a1[r];
    }
  }
  __syncthreads();
  if (pair == 1) {
    const int jcol = jt * 16 + jl;
    const float bi = bias4[0 * HP + jcol], bf = bias4[1 * HP + jcol];
    const float bg = bias4[2 * HP + jcol], bo = bias4[3 * HP + jcol];
    for (int r = 0; r < 8; ++r) {
      const int b = mt * 16 + r + hoff;
      float iv = sigm(ifl[((mt * 2 + 0) * 16 + r + hoff) * 16 + jl] + bi);
      float fv = sigm(ifl[((mt * 2 + 1) * 16 + r + hoff) * 16 + jl] + bf);
      float gv = tanhf(a0[r] + bg);
      float ov = sigm(a1[r] + bo);
      float cp = dn[b] ? 0.0f : cloc[r];
      float cn = fv * cp + iv * gv;
      float hn = ov * tanhf(cn);
      cloc[r] = cn;
      __bf16 hb16 = (__bf16)hn;
      hdst[(size_t)b * HP + jcol] = hb16;
      if (ydst) ydst[(size_t)b * HP + jcol] = hb16;
      if (hfin && jcol < HDIM) {
        hfin[b * HDIM + jcol] = hn;
        cfin[b * HDIM + jcol] = cn;
      }
    }
  }
}

__global__ __launch_bounds__(NTHREADS)
void k_core(const __bf16* __restrict__ xbuf,
            const unsigned char* __restrict__ done,
            const __bf16* __restrict__ wpack,
            const float* __restrict__ bias,      // [2][4][HP]
            __bf16* h0buf, __bf16* h1buf,        // [2 parities][BATCH][HP]
            __bf16* __restrict__ ybuf,           // [T][BATCH][HP]
            const float* __restrict__ c0in,      // [2][BATCH][HDIM]
            float* __restrict__ dout,
            unsigned* bar) {
  extern __shared__ char smem[];
  __bf16* wlds = (__bf16*)smem;                          // [2][4][KS2][512]  304 KB
  float*  ifl  = (float*)(smem + (size_t)2 * 4 * KS2 * 512 * sizeof(__bf16)); // 8 KB

  const int tid = threadIdx.x;
  const int jt = blockIdx.x;
  const int wave = tid >> 5;
  const int lane = tid & 31;

  // persistent weight slice -> LDS (both layers, this j-tile)
  {
    const uint4* src = (const uint4*)wpack;
    uint4* dst = (uint4*)wlds;
    const int fragU4 = 64;                                // 1 KB per fragment
    const int totalU4 = 2 * 4 * KS2 * fragU4;
    for (int i = tid; i < totalU4; i += NTHREADS) {
      int lf = i / fragU4;
      int wi = i - lf * fragU4;
      int ks = lf % KS2;
      int g  = (lf / KS2) & 3;
      int l  = lf / (KS2 * 4);
      size_t gf = ((size_t)(l * JT + jt) * 4 + g) * KS2 + ks;
      dst[i] = src[gf * fragU4 + wi];
    }
  }
  __syncthreads();

  // per-thread cell state for the owned (b, j) elements (g/o waves)
  float c0loc[8], c1loc[8];
  {
    const int mt = wave >> 1, pair = wave & 1;
    const int hl = lane >> 4, jl = lane & 15;
    const int jcol = jt * 16 + jl;
    for (int r = 0; r < 8; ++r) {
      int b = mt * 16 + r + 8 * hl;
      float v0 = 0.f, v1 = 0.f;
      if (pair == 1 && jcol < HDIM) {
        v0 = c0in[(size_t)(0 * BATCH + b) * HDIM + jcol];
        v1 = c0in[(size_t)(1 * BATCH + b) * HDIM + jcol];
      }
      c0loc[r] = v0; c1loc[r] = v1;
    }
  }

  const size_t hstride = (size_t)BATCH * HP;
  float* hT = dout + HT_OFF;
  float* cT = dout + CT_OFF;

  for (int t = 0; t < TSTEPS; ++t) {
    const int pin = t & 1, pout = pin ^ 1;
    const unsigned char* dn = done + (size_t)t * BATCH;
    const bool last = (t == TSTEPS - 1);

    // layer 0:  x = core_in[t],  h = h0[pin]
    lstm_phase(xbuf + (size_t)t * hstride, h0buf + (size_t)pin * hstride, dn,
               wlds, ifl, bias, c0loc,
               h0buf + (size_t)pout * hstride, nullptr,
               last ? hT : nullptr, last ? cT : nullptr,
               jt, wave, lane);
    grid_sync(bar);

    // layer 1:  x = h0_new,  h = h1[pin]
    lstm_phase(h0buf + (size_t)pout * hstride, h1buf + (size_t)pin * hstride, dn,
               wlds + (size_t)4 * KS2 * 512, ifl, bias + 4 * HP, c1loc,
               h1buf + (size_t)pout * hstride, ybuf + (size_t)t * hstride,
               last ? hT + BATCH * HDIM : nullptr, last ? cT + BATCH * HDIM : nullptr,
               jt, wave, lane);
    grid_sync(bar);
  }
}

// ---------------------------------------------------------------- heads
__global__ __launch_bounds__(NTHREADS)
void k_logits(const __bf16* __restrict__ ybuf,
              const __bf16* __restrict__ ppack,
              const float* __restrict__ policy_b,
              float* __restrict__ dout) {
  extern __shared__ char smem[];
  __bf16* pl = (__bf16*)smem;                  // [4][KSX][512] fragments, 76 KB
  const int tid = threadIdx.x;
  {
    const uint4* src = (const uint4*)ppack;
    uint4* dst = (uint4*)pl;
    for (int i = tid; i < 4 * KSX * 64; i += NTHREADS) dst[i] = src[i];
  }
  __syncthreads();

  const int wave = tid >> 5, lane = tid & 31;
  const int hl = lane >> 4, ml = lane & 15, jl = ml;
  const int mtile = blockIdx.x * 8 + wave;     // 2048 m-tiles total
  const int m0 = mtile * 16;

  v8f zero = {};
  v8f acc[4]; for (int n = 0; n < 4; ++n) acc[n] = zero;
  Frag fa, fb;
  for (int ks = 0; ks < KSX; ++ks) {
    const __bf16* row = ybuf + (size_t)(m0 + ml) * HP + ks * 32;
    fa.u[0] = *(const uint4*)(row + 8 * hl);
    fa.u[1] = *(const uint4*)(row + 16 + 8 * hl);
    for (int nt = 0; nt < 4; ++nt) {
      const __bf16* w = pl + ((size_t)(nt * KSX + ks)) * 512 + lane * 16;
      fb.u[0] = *(const uint4*)w; fb.u[1] = *(const uint4*)(w + 8);
      acc[nt] = wmma_bf16(fa, fb, acc[nt]);
    }
  }
  for (int nt = 0; nt < 4; ++nt) {
    const int col = nt * 16 + jl;
    const float pb = policy_b[col];
    for (int r = 0; r < 8; ++r) {
      const int m = m0 + r + 8 * hl;
      dout[(size_t)m * NACT + col] = acc[nt][r] + pb;
    }
  }
}

__global__ void k_tail(const __bf16* __restrict__ ybuf,
                       const float* __restrict__ bw,
                       const float* __restrict__ bb,
                       float* __restrict__ dout) {
  const int r = blockIdx.x * blockDim.x + threadIdx.x;
  if (r >= TSTEPS * BATCH) return;
  const __bf16* row = ybuf + (size_t)r * HP;
  float s = 0.f;
  for (int k = 0; k < HDIM; ++k) s += (float)row[k] * bw[k];
  dout[BASE_OFF + r] = s + bb[0];
  const float* lg = dout + (size_t)r * NACT;
  float best = lg[0]; int bi = 0;
  for (int n = 1; n < NACT; ++n) { float v = lg[n]; if (v > best) { best = v; bi = n; } }
  dout[ACT_OFF + r] = (float)bi;
}

// ---------------------------------------------------------------- launch
extern "C" void kernel_launch(void* const* d_in, const int* in_sizes, int n_in,
                              void* d_out, int out_size, void* d_ws, size_t ws_size,
                              hipStream_t stream) {
  (void)in_sizes; (void)n_in; (void)out_size; (void)ws_size;
  const float*         frame  = (const float*)d_in[0];
  const int*           act    = (const int*)d_in[1];
  const float*         rew    = (const float*)d_in[2];
  const unsigned char* done   = (const unsigned char*)d_in[3];  // jax bool
  const float*         h0in   = (const float*)d_in[4];
  const float*         c0in   = (const float*)d_in[5];
  const float*         wih0   = (const float*)d_in[6];
  const float*         whh0   = (const float*)d_in[7];
  const float*         wih1   = (const float*)d_in[8];
  const float*         whh1   = (const float*)d_in[9];
  const float*         bih0   = (const float*)d_in[10];
  const float*         bhh0   = (const float*)d_in[11];
  const float*         bih1   = (const float*)d_in[12];
  const float*         bhh1   = (const float*)d_in[13];
  const float*         pw     = (const float*)d_in[14];
  const float*         pb     = (const float*)d_in[15];
  const float*         blw    = (const float*)d_in[16];
  const float*         blb    = (const float*)d_in[17];
  float* dout = (float*)d_out;

  // workspace carve-up
  char* ws = (char*)d_ws;
  unsigned* bar  = (unsigned*)ws;
  __bf16* xbuf   = (__bf16*)(ws + 256);
  __bf16* ybuf   = xbuf + (size_t)TSTEPS * BATCH * HP;
  __bf16* wpack  = ybuf + (size_t)TSTEPS * BATCH * HP;
  __bf16* ppack  = wpack + (size_t)2 * JT * 4 * KS2 * 512;
  float*  bias   = (float*)(ppack + (size_t)4 * KSX * 512);
  __bf16* h0buf  = (__bf16*)(bias + 2 * 4 * HP);
  __bf16* h1buf  = h0buf + (size_t)2 * BATCH * HP;

  dim3 blk(NTHREADS);
  k_init<<<1, 64, 0, stream>>>(bar);
  {
    size_t n = (size_t)TSTEPS * BATCH * HP;                    // 19,922,944
    k_build_x<<<(unsigned)((n + NTHREADS - 1) / NTHREADS), blk, 0, stream>>>(frame, act, rew, xbuf);
  }
  {
    size_t n = (size_t)2 * JT * 4 * KS2 * 512;                 // 5,914,624
    k_pack_w<<<(unsigned)((n + NTHREADS - 1) / NTHREADS), blk, 0, stream>>>(wih0, whh0, wih1, whh1, wpack);
  }
  {
    size_t n = (size_t)4 * KSX * 512;                          // 38,912
    k_pack_p<<<(unsigned)((n + NTHREADS - 1) / NTHREADS), blk, 0, stream>>>(pw, ppack);
  }
  k_pack_b<<<(2 * 4 * HP + NTHREADS - 1) / NTHREADS, blk, 0, stream>>>(bih0, bhh0, bih1, bhh1, bias);
  k_init_h<<<(2 * BATCH * HP + NTHREADS - 1) / NTHREADS, blk, 0, stream>>>(h0in, h0buf, h1buf);

  const size_t ldsCore  = (size_t)2 * 4 * KS2 * 512 * sizeof(__bf16) + 4 * 2 * 16 * 16 * sizeof(float); // 319,488 B
  const size_t ldsHeads = (size_t)4 * KSX * 512 * sizeof(__bf16);                                       //  77,824 B

  k_core<<<NWG, blk, (unsigned)ldsCore, stream>>>(xbuf, done, wpack, bias,
                                                  h0buf, h1buf, ybuf, c0in, dout, bar);
  k_logits<<<(TSTEPS * BATCH / 16) / 8, blk, (unsigned)ldsHeads, stream>>>(ybuf, ppack, pb, dout);
  k_tail<<<(TSTEPS * BATCH) / NTHREADS, blk, 0, stream>>>(ybuf, blw, blb, dout);
}
